// Net_48902497632926
// MI455X (gfx1250) — matrix operand order repaired and checked by reading, not compiled, not run on hardware
//
#include <hip/hip_runtime.h>

typedef __attribute__((ext_vector_type(16))) _Float16 v16h;
typedef __attribute__((ext_vector_type(8)))  _Float16 v8h;
typedef __attribute__((ext_vector_type(8)))  float    v8f;

#define BN_EPS    1e-5f
#define NEG_SLOPE 0.01f
#define NBATCH    16
#define KNBR      64
#define BIGF      3.4e38f

// ---------------------------------------------------------------------------
// Weight prep: fold BN into W/b, pad (K->KPAD, N->NPAD), and swizzle W into
// WMMA B-fragment order so each lane loads its whole fragment as 32 contiguous bytes.
// Swizzled index: t = (((ks*NTILES)+nt)*32 + lane)*16 + e, where for lane
// (h=lane>>4, l16=lane&15): K = ks*32 + (e<8 ? h*8+e : 16+h*8+(e-8)), N = nt*16 + l16.
// ---------------------------------------------------------------------------
__global__ void prep_weights_kernel(const float* __restrict__ W, const float* __restrict__ bias,
                                    const float* __restrict__ gamma, const float* __restrict__ beta,
                                    _Float16* __restrict__ Wout, float* __restrict__ bout,
                                    int K, int N, int KPAD, int NPAD)
{
  const int ntiles = NPAD / 16;
  const int total  = KPAD * NPAD;
  const float sc = rsqrtf(1.0f + BN_EPS);
  for (int t = blockIdx.x * blockDim.x + threadIdx.x; t < total; t += gridDim.x * blockDim.x) {
    int e    = t & 15;
    int lane = (t >> 4) & 31;
    int grp  = t >> 9;               // ks*ntiles + nt
    int nt   = grp % ntiles;
    int ks   = grp / ntiles;
    int h    = lane >> 4, l16 = lane & 15;
    int k    = ks * 32 + ((e < 8) ? (h * 8 + e) : (16 + h * 8 + (e - 8)));
    int n    = nt * 16 + l16;
    float w = 0.f;
    if (k < K && n < N) {
      float s = gamma ? gamma[n] * sc : 1.f;
      w = W[k * N + n] * s;
    }
    Wout[t] = (_Float16)w;
  }
  for (int n = blockIdx.x * blockDim.x + threadIdx.x; n < NPAD; n += gridDim.x * blockDim.x) {
    float v = 0.f;
    if (n < N) {
      float s = gamma ? gamma[n] * sc : 1.f;
      v = bias[n] * s + (beta ? beta[n] : 0.f);
    }
    bout[n] = v;
  }
}

// ---------------------------------------------------------------------------
// Farthest point sampling: one workgroup per batch, distance array in LDS,
// sequential argmax chain (argmax ties -> lowest index, matching jnp.argmax).
// ---------------------------------------------------------------------------
template<int NPTS, int NSAMP>
__global__ __launch_bounds__(256) void fps_kernel(const float* __restrict__ pos,
                                                  int* __restrict__ outIdx)
{
  const int b = blockIdx.x;
  pos    += (size_t)b * NPTS * 3;
  outIdx += (size_t)b * NSAMP;
  __shared__ float dist[NPTS];
  __shared__ float rv[256];
  __shared__ int   ri[256];
  __shared__ int   lastS;
  const int tid = threadIdx.x;
  for (int i = tid; i < NPTS; i += 256) dist[i] = BIGF;
  if (tid == 0) lastS = 0;
  __syncthreads();
  for (int s = 0; s < NSAMP; ++s) {
    const int last = lastS;
    if (tid == 0) outIdx[s] = last;
    const float px = pos[last * 3], py = pos[last * 3 + 1], pz = pos[last * 3 + 2];
    float bv = -1.f; int bi = 0;
    for (int i = tid; i < NPTS; i += 256) {
      float dx = pos[i * 3] - px, dy = pos[i * 3 + 1] - py, dz = pos[i * 3 + 2] - pz;
      float d  = dx * dx + dy * dy + dz * dz;
      float dm = fminf(dist[i], d);
      dist[i] = dm;
      if (dm > bv) { bv = dm; bi = i; }
    }
    rv[tid] = bv; ri[tid] = bi;
    __syncthreads();
    for (int st = 128; st > 0; st >>= 1) {
      if (tid < st) {
        if (rv[tid + st] > rv[tid] || (rv[tid + st] == rv[tid] && ri[tid + st] < ri[tid])) {
          rv[tid] = rv[tid + st]; ri[tid] = ri[tid + st];
        }
      }
      __syncthreads();
    }
    if (tid == 0) lastS = ri[0];
    __syncthreads();
  }
}

__global__ void gather_pos_kernel(const float* __restrict__ pos, const int* __restrict__ idx,
                                  float* __restrict__ out, int N, int S)
{
  int t = blockIdx.x * blockDim.x + threadIdx.x;
  int total = NBATCH * S * 3;
  if (t >= total) return;
  int b = t / (S * 3);
  int r = t % (S * 3);
  int s = r / 3, c = r % 3;
  out[t] = pos[((size_t)b * N + idx[b * S + s]) * 3 + c];
}

// ---------------------------------------------------------------------------
// Radius top-k: one workgroup per destination point; d2 array in LDS;
// iteratively extract the 64 smallest d2 <= r^2 (== masked top_k(-d2, 64)).
// Invalid slots written as -1.
// ---------------------------------------------------------------------------
template<int NSRC>
__global__ __launch_bounds__(128) void radius_kernel(const float* __restrict__ posSrc,
                                                     const float* __restrict__ posDst,
                                                     int* __restrict__ nbr, float r2, int S)
{
  const int b = blockIdx.x / S;
  const int s = blockIdx.x % S;
  posSrc += (size_t)b * NSRC * 3;
  const float* pd = posDst + ((size_t)b * S + s) * 3;
  nbr += ((size_t)b * S + s) * KNBR;
  __shared__ float d2[NSRC];
  __shared__ float rv[128];
  __shared__ int   ri[128];
  const int tid = threadIdx.x;
  const float px = pd[0], py = pd[1], pz = pd[2];
  for (int i = tid; i < NSRC; i += 128) {
    float dx = posSrc[i * 3] - px, dy = posSrc[i * 3 + 1] - py, dz = posSrc[i * 3 + 2] - pz;
    float d = dx * dx + dy * dy + dz * dz;
    d2[i] = (d <= r2) ? d : BIGF;
  }
  __syncthreads();
  for (int k = 0; k < KNBR; ++k) {
    float bv = BIGF; int bi = NSRC;
    for (int i = tid; i < NSRC; i += 128) {
      float v = d2[i];
      if (v < bv) { bv = v; bi = i; }
    }
    rv[tid] = bv; ri[tid] = bi;
    __syncthreads();
    for (int st = 64; st > 0; st >>= 1) {
      if (tid < st) {
        if (rv[tid + st] < rv[tid] || (rv[tid + st] == rv[tid] && ri[tid + st] < ri[tid])) {
          rv[tid] = rv[tid + st]; ri[tid] = ri[tid + st];
        }
      }
      __syncthreads();
    }
    if (tid == 0) {
      int sel = (rv[0] < 3.3e38f) ? ri[0] : -1;
      nbr[k] = sel;
      if (sel >= 0) d2[sel] = BIGF;
    }
    __syncthreads();
  }
}

// ---------------------------------------------------------------------------
// WMMA MLP layers. Activations in LDS, weights pre-swizzled in global.
// A-fragment for lane (h,l16): two contiguous 16B chunks at k-offsets
// h*8 and 16+h*8 of row l16 (per ISA 16-bit A layout).
// N-tiles processed in groups of <=8 sharing one A-fragment per k-step:
// cuts LDS A-traffic up to 8x and issues back-to-back WMMAs (XDL-friendly).
// ---------------------------------------------------------------------------
template<int KPAD>
__device__ inline v16h load_a_frag(const _Float16* __restrict__ actIn,
                                   int rowBase, int l16, int h, int ks)
{
  const _Float16* pa = actIn + (rowBase + l16) * KPAD + ks * 32 + h * 8;
  v8h lo = *(const v8h*)pa;
  v8h hi = *(const v8h*)(pa + 16);
  v16h a;
#pragma unroll
  for (int e = 0; e < 8; ++e) { a[e] = lo[e]; a[e + 8] = hi[e]; }
  return a;
}

template<int ROWS, int KPAD, int NOUT>
__device__ inline void mlp_layer_act(const _Float16* __restrict__ actIn,
                                     const _Float16* __restrict__ Wf,
                                     const float* __restrict__ bias,
                                     _Float16* __restrict__ actOut)
{
  constexpr int KSTEPS = KPAD / 32;
  constexpr int NTILES = NOUT / 16;
  constexpr int NTG    = (NTILES < 8) ? NTILES : 8;
  const int lane = threadIdx.x & 31;
  const int wave = threadIdx.x >> 5;
  const int h = lane >> 4, l16 = lane & 15;
  const int rowBase = wave * 16;
  const v16h* Wv = (const v16h*)Wf;
  for (int nt0 = 0; nt0 < NTILES; nt0 += NTG) {
    v8f c[NTG];
#pragma unroll
    for (int j = 0; j < NTG; ++j)
#pragma unroll
      for (int i = 0; i < 8; ++i) c[j][i] = 0.f;
    for (int ks = 0; ks < KSTEPS; ++ks) {
      v16h a = load_a_frag<KPAD>(actIn, rowBase, l16, h, ks);
#pragma unroll
      for (int j = 0; j < NTG; ++j) {
        v16h bf = Wv[(ks * NTILES + nt0 + j) * 32 + lane];
        c[j] = __builtin_amdgcn_wmma_f32_16x16x32_f16(false, a, false, bf, (short)0, c[j],
                                                      false, false);
      }
    }
#pragma unroll
    for (int j = 0; j < NTG; ++j) {
      const int col = (nt0 + j) * 16 + l16;
      const float bb = bias[col];
#pragma unroll
      for (int r = 0; r < 8; ++r) {
        const int row = rowBase + h * 8 + r;
        float y = c[j][r] + bb;
        y = (y > 0.f) ? y : NEG_SLOPE * y;
        actOut[row * NOUT + col] = (_Float16)y;
      }
    }
  }
}

template<int ROWS, int KPAD, int NOUT>
__device__ inline void mlp_layer_final_max(const _Float16* __restrict__ actIn,
                                           const _Float16* __restrict__ Wf,
                                           const float* __restrict__ bias,
                                           const int* __restrict__ rowValid,
                                           float* __restrict__ partials)
{
  constexpr int KSTEPS = KPAD / 32;
  constexpr int NTILES = NOUT / 16;
  constexpr int NTG    = (NTILES < 8) ? NTILES : 8;
  const int lane = threadIdx.x & 31;
  const int wave = threadIdx.x >> 5;
  const int h = lane >> 4, l16 = lane & 15;
  const int rowBase = wave * 16;
  const v16h* Wv = (const v16h*)Wf;
  for (int nt0 = 0; nt0 < NTILES; nt0 += NTG) {
    v8f c[NTG];
#pragma unroll
    for (int j = 0; j < NTG; ++j)
#pragma unroll
      for (int i = 0; i < 8; ++i) c[j][i] = 0.f;
    for (int ks = 0; ks < KSTEPS; ++ks) {
      v16h a = load_a_frag<KPAD>(actIn, rowBase, l16, h, ks);
#pragma unroll
      for (int j = 0; j < NTG; ++j) {
        v16h bf = Wv[(ks * NTILES + nt0 + j) * 32 + lane];
        c[j] = __builtin_amdgcn_wmma_f32_16x16x32_f16(false, a, false, bf, (short)0, c[j],
                                                      false, false);
      }
    }
#pragma unroll
    for (int j = 0; j < NTG; ++j) {
      const int col = (nt0 + j) * 16 + l16;
      const float bb = bias[col];
      float mx = -BIGF;
#pragma unroll
      for (int r = 0; r < 8; ++r) {
        const int row = rowBase + h * 8 + r;
        if (rowValid[row]) mx = fmaxf(mx, c[j][r] + bb);
      }
      partials[(wave * 2 + h) * NOUT + col] = mx;
    }
  }
}

// Cover the first-layer weights with prefetches while staging runs.
__device__ inline void prefetch_weights(const _Float16* __restrict__ w, int bytes,
                                        const _Float16* __restrict__ wn1,
                                        const _Float16* __restrict__ wn2)
{
  const int tid = threadIdx.x;
  const int lines = bytes >> 8;
  for (int p = tid; p < lines; p += blockDim.x)
    __builtin_prefetch((const char*)w + (p << 8), 0, 3);
  if (tid == 0) {
    __builtin_prefetch(wn1, 0, 2);
    __builtin_prefetch(wn2, 0, 2);
  }
}

// ---------------------------------------------------------------------------
// Fused PointNetConv: one workgroup (128 thr = 4 waves) per (batch, dst point).
// Gather 64 neighbors (features + rel pos) -> 3-layer MLP (WMMA) -> masked max.
// ---------------------------------------------------------------------------
template<int CIN, int CINPAD, int C1, int C2, int C3, int B0>
__global__ __launch_bounds__(128) void conv_kernel(
    const float* __restrict__ xsrc, int F, int Nsrc,
    const float* __restrict__ posSrc, const float* __restrict__ posDst,
    const int* __restrict__ nbr, int S,
    const _Float16* __restrict__ w1, const float* __restrict__ b1,
    const _Float16* __restrict__ w2, const float* __restrict__ b2,
    const _Float16* __restrict__ w3, const float* __restrict__ b3,
    float* __restrict__ out)
{
  constexpr int ROWS = 64;
  const int b = blockIdx.x / S;
  const int s = blockIdx.x % S;
  __shared__ alignas(16) _Float16 buf0[ROWS * B0];
  __shared__ alignas(16) _Float16 buf1[ROWS * C1];
  __shared__ float partials[8 * C3];
  __shared__ int rowIdx[ROWS];
  __shared__ int rowValid[ROWS];
  const int tid = threadIdx.x;
  prefetch_weights(w1, CINPAD * C1 * 2, w2, w3);
  if (tid < ROWS) {
    int id = nbr[((size_t)b * S + s) * KNBR + tid];
    rowIdx[tid] = id;
    rowValid[tid] = (id >= 0);
  }
  __syncthreads();
  const float* pd = posDst + ((size_t)b * S + s) * 3;
  const float px = pd[0], py = pd[1], pz = pd[2];
  for (int t = tid; t < ROWS * CINPAD; t += 128) {
    int row = t / CINPAD, c = t % CINPAD;
    float v = 0.f;
    int id = rowIdx[row];
    if (id >= 0) {
      if (c < F) v = xsrc[((size_t)b * Nsrc + id) * F + c];
      else if (c < CIN) {
        float dc = posSrc[((size_t)b * Nsrc + id) * 3 + (c - F)];
        v = dc - ((c - F) == 0 ? px : ((c - F) == 1 ? py : pz));
      }
    }
    buf0[row * CINPAD + c] = (_Float16)v;
  }
  __syncthreads();
  mlp_layer_act<ROWS, CINPAD, C1>(buf0, w1, b1, buf1);
  __syncthreads();
  mlp_layer_act<ROWS, C1, C2>(buf1, w2, b2, buf0);
  __syncthreads();
  mlp_layer_final_max<ROWS, C2, C3>(buf0, w3, b3, rowValid, partials);
  __syncthreads();
  for (int c = tid; c < C3; c += 128) {
    float m = partials[c];
#pragma unroll
    for (int p = 1; p < 8; ++p) m = fmaxf(m, partials[p * C3 + c]);
    out[((size_t)b * S + s) * C3 + c] = m;
  }
}

// ---------------------------------------------------------------------------
// SA3: per-point MLP over 16-point chunks + partial global max per chunk.
// ---------------------------------------------------------------------------
template<int CINPAD, int C1, int C2, int C3, int B0>
__global__ __launch_bounds__(32) void sa3_kernel(
    const float* __restrict__ xin, const float* __restrict__ posin,
    int S, int F,
    const _Float16* __restrict__ w1, const float* __restrict__ b1,
    const _Float16* __restrict__ w2, const float* __restrict__ b2,
    const _Float16* __restrict__ w3, const float* __restrict__ b3,
    float* __restrict__ part, int chunks)
{
  constexpr int ROWS = 16;
  const int b  = blockIdx.x / chunks;
  const int ch = blockIdx.x % chunks;
  const int row0 = ch * ROWS;
  __shared__ alignas(16) _Float16 buf0[ROWS * B0];
  __shared__ alignas(16) _Float16 buf1[ROWS * C1];
  __shared__ float partials[2 * C3];
  __shared__ int rowValid[ROWS];
  const int tid = threadIdx.x;
  prefetch_weights(w1, CINPAD * C1 * 2, w2, w3);
  if (tid < ROWS) rowValid[tid] = (row0 + tid) < S;
  for (int t = tid; t < ROWS * CINPAD; t += 32) {
    int row = t / CINPAD, c = t % CINPAD;
    int sidx = row0 + row;
    float v = 0.f;
    if (sidx < S) {
      if (c < F) v = xin[((size_t)b * S + sidx) * F + c];
      else if (c < F + 3) v = posin[((size_t)b * S + sidx) * 3 + (c - F)];
    }
    buf0[row * CINPAD + c] = (_Float16)v;
  }
  __syncthreads();
  mlp_layer_act<ROWS, CINPAD, C1>(buf0, w1, b1, buf1);
  __syncthreads();
  mlp_layer_act<ROWS, C1, C2>(buf1, w2, b2, buf0);
  __syncthreads();
  mlp_layer_final_max<ROWS, C2, C3>(buf0, w3, b3, rowValid, partials);
  __syncthreads();
  for (int c = tid; c < C3; c += 32)
    part[((size_t)b * chunks + ch) * C3 + c] = fmaxf(partials[c], partials[C3 + c]);
}

__global__ void reduce_max_kernel(const float* __restrict__ part, float* __restrict__ g, int chunks)
{
  int t = blockIdx.x * blockDim.x + threadIdx.x;
  if (t >= NBATCH * 1024) return;
  int b = t / 1024, c = t % 1024;
  float m = part[((size_t)b * chunks) * 1024 + c];
  for (int p = 1; p < chunks; ++p) m = fmaxf(m, part[((size_t)b * chunks + p) * 1024 + c]);
  g[t] = m;
}

// ---------------------------------------------------------------------------
// Head MLP: 16 rows (batches), single wave. Last layer padded to 16 cols.
// ---------------------------------------------------------------------------
template<int K0, int C1, int C2>
__global__ __launch_bounds__(32) void head_kernel(
    const float* __restrict__ g,
    const _Float16* __restrict__ w1, const float* __restrict__ b1,
    const _Float16* __restrict__ w2, const float* __restrict__ b2,
    const _Float16* __restrict__ w3, const float* __restrict__ b3,
    float* __restrict__ out)
{
  constexpr int ROWS = 16;
  __shared__ alignas(16) _Float16 buf0[ROWS * K0];
  __shared__ alignas(16) _Float16 buf1[ROWS * C1];
  const int tid = threadIdx.x;
  prefetch_weights(w1, K0 * C1 * 2, w2, w3);
  for (int t = tid; t < ROWS * K0; t += 32) buf0[t] = (_Float16)g[t];
  __syncthreads();
  mlp_layer_act<ROWS, K0, C1>(buf0, w1, b1, buf1);
  __syncthreads();
  mlp_layer_act<ROWS, C1, C2>(buf1, w2, b2, buf0);
  __syncthreads();
  {
    const v16h* Wv = (const v16h*)w3;
    const int lane = tid & 31;
    const int h = lane >> 4, l16 = lane & 15;
    v8f c;
#pragma unroll
    for (int i = 0; i < 8; ++i) c[i] = 0.f;
    for (int ks = 0; ks < C2 / 32; ++ks) {
      v16h a = load_a_frag<C2>(buf0, 0, l16, h, ks);
      v16h bf = Wv[ks * 32 + lane];
      c = __builtin_amdgcn_wmma_f32_16x16x32_f16(false, a, false, bf, (short)0, c, false, false);
    }
    if (l16 == 0) {
      const float bb = b3[0];
#pragma unroll
      for (int r = 0; r < 8; ++r) out[h * 8 + r] = c[r] + bb;
    }
  }
}

// ---------------------------------------------------------------------------
extern "C" void kernel_launch(void* const* d_in, const int* in_sizes, int n_in,
                              void* d_out, int out_size, void* d_ws, size_t ws_size,
                              hipStream_t stream) {
  (void)in_sizes; (void)n_in; (void)out_size; (void)ws_size;
  const float* x   = (const float*)d_in[0];
  const float* pos = (const float*)d_in[1];

  char* ws = (char*)d_ws;
  size_t off = 0;
  auto alloc = [&](size_t bytes) -> void* {
    void* p = ws + off;
    off += (bytes + 255) & ~(size_t)255;
    return p;
  };

  const int S1v = 819, S2v = 204, CH3 = 13;

  int*   fps1  = (int*)  alloc((size_t)NBATCH * S1v * 4);
  float* pos1  = (float*)alloc((size_t)NBATCH * S1v * 3 * 4);
  int*   nbr1  = (int*)  alloc((size_t)NBATCH * S1v * KNBR * 4);
  float* x1    = (float*)alloc((size_t)NBATCH * S1v * 128 * 4);
  int*   fps2  = (int*)  alloc((size_t)NBATCH * S2v * 4);
  float* pos2  = (float*)alloc((size_t)NBATCH * S2v * 3 * 4);
  int*   nbr2  = (int*)  alloc((size_t)NBATCH * S2v * KNBR * 4);
  float* x2    = (float*)alloc((size_t)NBATCH * S2v * 256 * 4);
  float* part  = (float*)alloc((size_t)NBATCH * CH3 * 1024 * 4);
  float* gfeat = (float*)alloc((size_t)NBATCH * 1024 * 4);

  struct Spec { int base; int K, N, KPAD, NPAD; int bn; };
  const Spec specs[12] = {
    {2,    7,   64,   32,   64, 1}, {6,   64,   64,   64,   64, 1}, {10,  64,  128,  64,  128, 0},
    {12, 131,  128,  160,  128, 1}, {16, 128,  128,  128,  128, 1}, {20, 128,  256, 128,  256, 0},
    {22, 259,  256,  288,  256, 1}, {26, 256,  512,  256,  512, 1}, {30, 512, 1024, 512, 1024, 0},
    {32, 1024, 256, 1024,  256, 1}, {36, 256,  256,  256,  256, 1}, {40, 256,    1, 256,   16, 0},
  };
  _Float16* Wp[12]; float* Bp[12];
  for (int i = 0; i < 12; ++i) {
    Wp[i] = (_Float16*)alloc((size_t)specs[i].KPAD * specs[i].NPAD * 2);
    Bp[i] = (float*)   alloc((size_t)specs[i].NPAD * 4);
  }
  for (int i = 0; i < 12; ++i) {
    const Spec& sp = specs[i];
    const float* W  = (const float*)d_in[sp.base];
    const float* bb = (const float*)d_in[sp.base + 1];
    const float* gg = sp.bn ? (const float*)d_in[sp.base + 2] : nullptr;
    const float* bt = sp.bn ? (const float*)d_in[sp.base + 3] : nullptr;
    int total = sp.KPAD * sp.NPAD;
    int grid = (total + 255) / 256; if (grid > 1024) grid = 1024;
    prep_weights_kernel<<<grid, 256, 0, stream>>>(W, bb, gg, bt, Wp[i], Bp[i],
                                                  sp.K, sp.N, sp.KPAD, sp.NPAD);
  }

  // ---- SA1 ----
  fps_kernel<4096, 819><<<NBATCH, 256, 0, stream>>>(pos, fps1);
  gather_pos_kernel<<<(NBATCH * S1v * 3 + 255) / 256, 256, 0, stream>>>(pos, fps1, pos1, 4096, S1v);
  radius_kernel<4096><<<NBATCH * S1v, 128, 0, stream>>>(pos, pos1, nbr1, 0.01f, S1v);
  conv_kernel<7, 32, 64, 64, 128, 64><<<NBATCH * S1v, 128, 0, stream>>>(
      x, 4, 4096, pos, pos1, nbr1, S1v, Wp[0], Bp[0], Wp[1], Bp[1], Wp[2], Bp[2], x1);

  // ---- SA2 ----
  fps_kernel<819, 204><<<NBATCH, 256, 0, stream>>>(pos1, fps2);
  gather_pos_kernel<<<(NBATCH * S2v * 3 + 255) / 256, 256, 0, stream>>>(pos1, fps2, pos2, 819, S2v);
  radius_kernel<819><<<NBATCH * S2v, 128, 0, stream>>>(pos1, pos2, nbr2, 0.25f, S2v);
  conv_kernel<131, 160, 128, 128, 256, 160><<<NBATCH * S2v, 128, 0, stream>>>(
      x1, 128, 819, pos1, pos2, nbr2, S2v, Wp[3], Bp[3], Wp[4], Bp[4], Wp[5], Bp[5], x2);

  // ---- SA3 (per-point MLP + global max) ----
  sa3_kernel<288, 256, 512, 1024, 512><<<NBATCH * CH3, 32, 0, stream>>>(
      x2, pos2, S2v, 256, Wp[6], Bp[6], Wp[7], Bp[7], Wp[8], Bp[8], part, CH3);
  reduce_max_kernel<<<(NBATCH * 1024 + 255) / 256, 256, 0, stream>>>(part, gfeat, CH3);

  // ---- Head ----
  head_kernel<1024, 256, 256><<<1, 32, 0, stream>>>(
      gfeat, Wp[9], Bp[9], Wp[10], Bp[10], Wp[11], Bp[11], (float*)d_out);
}
